// OIMLossNorm_9105330667996
// MI455X (gfx1250) — compile-verified, matrix operand order checked by reading, not run on hardware
//
#include <hip/hip_runtime.h>
#include <hip/hip_bf16.h>
#include <math.h>

typedef __attribute__((ext_vector_type(16))) __bf16 v16bf;
typedef __attribute__((ext_vector_type(8)))  __bf16 v8bf;
typedef __attribute__((ext_vector_type(8)))  float  v8f;

#define FEAT   256
#define BATCH  2048
#define NPID   30000
#define CQN    8192
#define NCOL   (NPID + CQN)        // 38192
#define NCT    (NCOL / 16)         // 2387 column tiles
#define NSLAB  64                  // column slabs
#define TPS    ((NCT + NSLAB - 1) / NSLAB)  // 38 tiles per slab
#define NRB    (BATCH / 256)       // 8 row blocks (16 row tiles each, 2 per wave)
#define NRT    (BATCH / 16)        // 128 row tiles
#define SCAL   30.0f
#define EPSBN  1e-5f
#define LDSPITCH 264               // 256 + 8 bf16 pad -> 528B row, bank-conflict free

// ---------------------------------------------------------------- bad masks
__global__ void kBadMask(const float* __restrict__ lut, const float* __restrict__ cq,
                         int* badLut, int* badCq) {
  int wave = (blockIdx.x * blockDim.x + threadIdx.x) >> 5;
  int lane = threadIdx.x & 31;
  if (wave >= NCOL) return;
  const float* row = (wave < NPID) ? (lut + (size_t)wave * FEAT)
                                   : (cq + (size_t)(wave - NPID) * FEAT);
  const float4* r4 = (const float4*)row;
  int nz = 0;
  for (int c = 0; c < 2; ++c) {
    float4 v = r4[lane * 2 + c];
    nz |= (v.x != 0.f) | (v.y != 0.f) | (v.z != 0.f) | (v.w != 0.f);
  }
  nz = __any(nz);
  if (lane == 0) {
    if (wave < NPID) badLut[wave] = !nz;
    else             badCq[wave - NPID] = !nz;
  }
}

// -------------------------------------------------- masked BN stats, pass 1
__global__ void kStats(const float* __restrict__ lut, const int* __restrict__ badLut,
                       float* part) {
  int f = threadIdx.x;            // feature
  int bl = blockIdx.x;            // 120 blocks x 250 rows
  int r0 = bl * 250, r1 = r0 + 250;
  float s = 0.f, s2 = 0.f;
  for (int r = r0; r < r1; ++r) {
    if (!badLut[r]) {
      float v = lut[(size_t)r * FEAT + f];
      s += v; s2 += v * v;
    }
  }
  part[(size_t)bl * 512 + f]       = s;
  part[(size_t)bl * 512 + 256 + f] = s2;
}

// -------------------------------------------------- masked BN stats, pass 2
__global__ void kStatsReduce(const float* __restrict__ part, const int* __restrict__ badLut,
                             const float* __restrict__ w, const float* __restrict__ b,
                             float* stats) {
  __shared__ float sh[256];
  __shared__ float nsh;
  int f = threadIdx.x;
  int cnt = 0;
  for (int i = f; i < NPID; i += 256) cnt += badLut[i];
  sh[f] = (float)cnt; __syncthreads();
  for (int o = 128; o > 0; o >>= 1) { if (f < o) sh[f] += sh[f + o]; __syncthreads(); }
  if (f == 0) nsh = (float)NPID - sh[0];
  __syncthreads();
  float n = nsh;
  float S = 0.f, S2 = 0.f;
  for (int bl = 0; bl < 120; ++bl) {
    S  += part[(size_t)bl * 512 + f];
    S2 += part[(size_t)bl * 512 + 256 + f];
  }
  float mean = S / n;
  float var  = S2 / n - mean * mean;
  if (var < 0.f) var = 0.f;
  float al  = w[f] * rsqrtf(var + EPSBN);           // train-mode (lut) scale
  float rv1 = var * n / (n - 1.0f);                 // unbiased running var
  float ac  = w[f] * rsqrtf(rv1 + EPSBN);           // eval-mode (cq) scale
  stats[        f] = al;
  stats[256 +   f] = b[f] - mean * al;
  stats[512 +   f] = ac;
  stats[768 +   f] = b[f] - mean * ac;
}

// ------------------------------------------------ qn = l2norm(BN(inputs)) -> bf16
__global__ void kQn(const float* __restrict__ inp, const float* __restrict__ w,
                    const float* __restrict__ b, __bf16* qn) {
  int wave = (blockIdx.x * blockDim.x + threadIdx.x) >> 5;
  int lane = threadIdx.x & 31;
  if (wave >= BATCH) return;
  const float inv = rsqrtf(1.0f + EPSBN);
  float x[8]; float ss = 0.f;
  int base = lane * 8;
  for (int j = 0; j < 8; ++j) {
    int f = base + j;
    float v = inp[(size_t)wave * FEAT + f] * (w[f] * inv) + b[f];
    x[j] = v; ss += v * v;
  }
  for (int o = 16; o > 0; o >>= 1) ss += __shfl_xor(ss, o, 32);
  float sc = 1.0f / fmaxf(sqrtf(ss), 1e-12f);
  for (int j = 0; j < 8; ++j) qn[(size_t)wave * FEAT + base + j] = (__bf16)(x[j] * sc);
}

// --------------------- W = [l2norm(lut_bn); l2norm(cq_bn)] -> bf16, + colBad
__global__ void kW(const float* __restrict__ lut, const float* __restrict__ cq,
                   const int* __restrict__ badLut, const int* __restrict__ badCq,
                   const float* __restrict__ stats, __bf16* W, int* colBad) {
  int wave = (blockIdx.x * blockDim.x + threadIdx.x) >> 5;
  int lane = threadIdx.x & 31;
  if (wave >= NCOL) return;
  bool isLut = wave < NPID;
  int bad = isLut ? badLut[wave] : badCq[wave - NPID];
  const float* src = isLut ? lut + (size_t)wave * FEAT : cq + (size_t)(wave - NPID) * FEAT;
  const float* A  = isLut ? stats       : stats + 512;
  const float* Bo = isLut ? stats + 256 : stats + 768;
  int base = lane * 8;
  if (bad) {  // bad row: raw row is all zeros -> l2norm gives zeros
    for (int j = 0; j < 8; ++j) W[(size_t)wave * FEAT + base + j] = (__bf16)0.0f;
    if (lane == 0) colBad[wave] = 1;
    return;
  }
  float x[8]; float ss = 0.f;
  for (int j = 0; j < 8; ++j) {
    int f = base + j;
    float v = src[f] * A[f] + Bo[f];
    x[j] = v; ss += v * v;
  }
  for (int o = 16; o > 0; o >>= 1) ss += __shfl_xor(ss, o, 32);
  float sc = 1.0f / fmaxf(sqrtf(ss), 1e-12f);
  for (int j = 0; j < 8; ++j) W[(size_t)wave * FEAT + base + j] = (__bf16)(x[j] * sc);
  if (lane == 0) colBad[wave] = 0;
}

// ---------------------- per-row aux: bpc = (bad-position ? lc : -1) -------
__global__ void kRowAux(const int* __restrict__ label, const int* __restrict__ badLut,
                        int* bpc) {
  int i = blockIdx.x * blockDim.x + threadIdx.x;
  if (i >= BATCH) return;
  int lab = label[i];
  int lc = (lab < NPID - 1) ? lab : (NPID - 1);
  bpc[i] = ((lab < NPID) && badLut[lc]) ? lc : -1;
}

// ---------------- fused bf16 WMMA GEMM + streaming softmax partial sums ----
// grid = NRB*NSLAB blocks of 256 (8 waves). Block covers 256 rows: each wave
// owns TWO 16-row tiles so every LDS B-fragment pair feeds two WMMAs
// (halves LDS bandwidth per WMMA). B tiles (16 cols x 256 K, 8KB) are staged
// into LDS double-buffered and shared by all 8 waves. A fragments are
// register-resident. Logits never materialized: epilogue accumulates
// sum(exp(logit-30)) per row (cosine*30 <= 30 -> fixed max).
__global__ __launch_bounds__(256)
void kGemm(const __bf16* __restrict__ Qn, const __bf16* __restrict__ W,
           const int* __restrict__ colBad, const int* __restrict__ bpc,
           float* __restrict__ sPart) {
  int rb   = blockIdx.x >> 6;      // row block: 0..7  (rows rb*256 .. +256)
  int slab = blockIdx.x & 63;      // column slab
  int wv   = threadIdx.x >> 5;     // wave
  int lane = threadIdx.x & 31;
  int m = lane & 15, half = lane >> 4;
  int rtA = rb * 16 + wv * 2;      // first row tile of this wave
  int rtB = rtA + 1;               // second row tile

  __shared__ __align__(16) __bf16 bbuf[2][16 * LDSPITCH];

  // Preload A fragments for both row tiles (2 x 16 rows x 256 K) into regs.
  // 16-bit A layout: lanes 0-15 hold K {k0+0..7, k0+16..23}; lanes 16-31 the rest.
  v16bf afA[8], afB[8];
  {
    const __bf16* arowA = Qn + (size_t)(rtA * 16 + m) * FEAT;
    const __bf16* arowB = Qn + (size_t)(rtB * 16 + m) * FEAT;
    for (int ks = 0; ks < 8; ++ks) {
      int k0 = ks * 32;
      v8bf a0 = *(const v8bf*)(arowA + k0 + half * 8);
      v8bf a1 = *(const v8bf*)(arowA + k0 + 16 + half * 8);
      v8bf b0 = *(const v8bf*)(arowB + k0 + half * 8);
      v8bf b1 = *(const v8bf*)(arowB + k0 + 16 + half * 8);
      v16bf a, b;
      for (int j = 0; j < 8; ++j) { a[j] = a0[j]; a[8 + j] = a1[j];
                                    b[j] = b0[j]; b[8 + j] = b1[j]; }
      afA[ks] = a; afB[ks] = b;
    }
  }
  // bad-position column (or -1) for this lane's 16 output rows
  // (C layout: row = half*8 + r, col = lane&15)
  int bpA_[8], bpB_[8];
  for (int r = 0; r < 8; ++r) {
    bpA_[r] = bpc[rtA * 16 + half * 8 + r];
    bpB_[r] = bpc[rtB * 16 + half * 8 + r];
  }
  float sA[8], sB[8];
  for (int r = 0; r < 8; ++r) { sA[r] = 0.f; sB[r] = 0.f; }

  int t0 = slab * TPS;
  int t1 = t0 + TPS; if (t1 > NCT) t1 = NCT;

  // cooperative copy of one B tile (16 cols x 256 K) into LDS buffer `pb`
  auto copyTile = [&](int pb, int t) {
    int gcol0 = t * 16;
    for (int c = threadIdx.x; c < 512; c += 256) {     // 512 chunks of 16B
      int col = c >> 5;
      int ko  = (c & 31) * 8;
      v8bf v = *(const v8bf*)(W + (size_t)(gcol0 + col) * FEAT + ko);
      *(v8bf*)&bbuf[pb][col * LDSPITCH + ko] = v;
    }
  };

  int pb = 0;
  if (t0 < t1) copyTile(0, t0);
  __syncthreads();

  for (int t = t0; t < t1; ++t) {
    if (t + 1 < t1) {
      copyTile(pb ^ 1, t + 1);
      if (t + 2 < t1)
        __builtin_prefetch(W + (size_t)(t + 2) * 16 * FEAT + threadIdx.x * 16, 0, 1);
    }
    int col = t * 16 + m;
    const __bf16* bcol = &bbuf[pb][m * LDSPITCH];
    v8f accA = {}, accB = {};
    // software-pipelined: fetch K-step ks+1 from LDS while WMMAs for ks issue
    v8bf c0 = *(const v8bf*)(bcol + half * 8);
    v8bf c1 = *(const v8bf*)(bcol + 16 + half * 8);
#pragma unroll
    for (int ks = 0; ks < 8; ++ks) {
      v8bf n0, n1;
      if (ks < 7) {
        int k0n = (ks + 1) * 32;
        n0 = *(const v8bf*)(bcol + k0n + half * 8);
        n1 = *(const v8bf*)(bcol + k0n + 16 + half * 8);
      }
      v16bf bb;
      for (int j = 0; j < 8; ++j) { bb[j] = c0[j]; bb[8 + j] = c1[j]; }
      accA = __builtin_amdgcn_wmma_f32_16x16x32_bf16(
          false, afA[ks], false, bb, (short)0, accA, false, false);
      accB = __builtin_amdgcn_wmma_f32_16x16x32_bf16(
          false, afB[ks], false, bb, (short)0, accB, false, false);
      c0 = n0; c1 = n1;
    }
    int cb = colBad[col];
    for (int r = 0; r < 8; ++r) {
      float vA = accA[r] * SCAL;
      float vB = accB[r] * SCAL;
      if (cb) { vA = -SCAL; vB = -SCAL; }        // bad column -> -1 * 30
      if (bpA_[r] == col) vA = SCAL;             // bad-position override -> +1 * 30
      if (bpB_[r] == col) vB = SCAL;
      sA[r] += __expf(vA - SCAL);                // fixed max = 30
      sB[r] += __expf(vB - SCAL);
    }
    __syncthreads();
    pb ^= 1;
  }

  // reduce across the 16 lanes that share rows (stays within half: offsets 1..8)
  for (int o = 1; o < 16; o <<= 1)
    for (int r = 0; r < 8; ++r) {
      sA[r] += __shfl_xor(sA[r], o, 32);
      sB[r] += __shfl_xor(sB[r], o, 32);
    }
  if (m == 0) {
    for (int r = 0; r < 8; ++r) {
      sPart[((size_t)rtA * NSLAB + slab) * 16 + half * 8 + r] = sA[r];
      sPart[((size_t)rtB * NSLAB + slab) * 16 + half * 8 + r] = sB[r];
    }
  }
}

// ----------------------- per-row: target logit + per-sample CE term -------
__global__ void kTargetRow(const __bf16* __restrict__ Qn, const __bf16* __restrict__ W,
                           const int* __restrict__ label, const int* __restrict__ bpc,
                           const float* __restrict__ sPart, float* per) {
  int wave = (blockIdx.x * blockDim.x + threadIdx.x) >> 5;
  int lane = threadIdx.x & 31;
  if (wave >= BATCH) return;
  int lab = label[wave];
  if (lab == NPID) { if (lane == 0) per[wave] = 0.f; return; }
  int rt = wave >> 4, ri = wave & 15;
  float s = 0.f;
  for (int sl = lane; sl < NSLAB; sl += 32)     // parallel over slabs
    s += sPart[((size_t)rt * NSLAB + sl) * 16 + ri];
  for (int o = 16; o > 0; o >>= 1) s += __shfl_xor(s, o, 32);
  float t;
  if (bpc[wave] >= 0) {
    t = SCAL;
  } else {
    float d = 0.f; int base = lane * 8;
    const __bf16* q  = Qn + (size_t)wave * FEAT;
    const __bf16* ww = W + (size_t)lab * FEAT;
    for (int j = 0; j < 8; ++j) d += (float)q[base + j] * (float)ww[base + j];
    for (int o = 16; o > 0; o >>= 1) d += __shfl_xor(d, o, 32);
    t = d * SCAL;
  }
  if (lane == 0) per[wave] = SCAL + logf(s) - t;   // logsumexp - target
}

__global__ void kLossReduce(const float* __restrict__ per, float* out) {
  __shared__ float sh[256];
  int t = threadIdx.x;
  float s = 0.f;
  for (int i = t; i < BATCH; i += 256) s += per[i];
  sh[t] = s; __syncthreads();
  for (int o = 128; o > 0; o >>= 1) { if (t < o) sh[t] += sh[t + o]; __syncthreads(); }
  if (t == 0) out[0] = sh[0] / (float)BATCH;
}

// ---------------- circular-queue positions (exclusive scan of unlabeled) --
__global__ void kCqScan(const int* __restrict__ label, int* cqpos) {
  __shared__ int sh[256];
  int t = threadIdx.x;
  int c = 0;
  for (int i = t * 8; i < t * 8 + 8; ++i) c += (label[i] == NPID);
  sh[t] = c; __syncthreads();
  for (int o = 1; o < 256; o <<= 1) {
    int v = (t >= o) ? sh[t - o] : 0;
    __syncthreads();
    sh[t] += v;
    __syncthreads();
  }
  int off = sh[t] - c;  // exclusive prefix
  for (int i = t * 8; i < t * 8 + 8; ++i) {
    if (label[i] == NPID) cqpos[i] = off++;
    else                  cqpos[i] = -1;
  }
}

__global__ void kCqWrite(const float* __restrict__ inp, const int* __restrict__ cqpos,
                         float* outCq) {
  int wave = (blockIdx.x * blockDim.x + threadIdx.x) >> 5;
  int lane = threadIdx.x & 31;
  if (wave >= BATCH) return;
  int p = cqpos[wave];
  if (p < 0) return;
  p = p % CQN;  // U <= 2048 < 8192, positions unique
  for (int j = lane; j < FEAT; j += 32)
    outCq[(size_t)p * FEAT + j] = inp[(size_t)wave * FEAT + j];
}

// --------- order-exact momentum scatter: last occurrence replays its chain -
__global__ void kLutUpdate(const float* __restrict__ inp, const int* __restrict__ label,
                           const float* __restrict__ lut, float* outLut) {
  int wave = (blockIdx.x * blockDim.x + threadIdx.x) >> 5;
  int lane = threadIdx.x & 31;
  if (wave >= BATCH) return;
  int lab = label[wave];
  if (lab >= NPID) return;
  for (int j = wave + 1; j < BATCH; ++j)
    if (label[j] == lab) return;           // only the last occurrence writes
  float acc[8]; int base = lane * 8;
  for (int e = 0; e < 8; ++e) acc[e] = lut[(size_t)lab * FEAT + base + e];
  for (int j = 0; j <= wave; ++j) {
    if (label[j] == lab) {
      for (int e = 0; e < 8; ++e)
        acc[e] = 0.5f * acc[e] + 0.5f * inp[(size_t)j * FEAT + base + e];
    }
  }
  for (int e = 0; e < 8; ++e) outLut[(size_t)lab * FEAT + base + e] = acc[e];
}

// --------------------------------------------------------------------------
extern "C" void kernel_launch(void* const* d_in, const int* in_sizes, int n_in,
                              void* d_out, int out_size, void* d_ws, size_t ws_size,
                              hipStream_t stream) {
  const float* inp   = (const float*)d_in[0];
  const int*   label = (const int*)  d_in[1];
  const float* lut   = (const float*)d_in[2];
  const float* cq    = (const float*)d_in[3];
  const float* bw    = (const float*)d_in[4];
  const float* bb    = (const float*)d_in[5];
  float* out = (float*)d_out;
  float* outLut = out + 1;
  float* outCq  = out + 1 + (size_t)NPID * FEAT;

  char* ws = (char*)d_ws;
  size_t off = 0;
  auto alloc = [&](size_t bytes) -> char* {
    char* p = ws + off;
    off = (off + bytes + 255) & ~(size_t)255;
    return p;
  };
  int*    badLut  = (int*)   alloc((size_t)NPID * 4);
  int*    badCq   = (int*)   alloc((size_t)CQN * 4);
  float*  part    = (float*) alloc((size_t)120 * 512 * 4);
  float*  stats   = (float*) alloc((size_t)4 * 256 * 4);
  __bf16* Qn      = (__bf16*)alloc((size_t)BATCH * FEAT * 2);
  __bf16* Wm      = (__bf16*)alloc((size_t)NCOL * FEAT * 2);
  int*    colBad  = (int*)   alloc((size_t)NCOL * 4);
  int*    bpc     = (int*)   alloc((size_t)BATCH * 4);
  float*  sPart   = (float*) alloc((size_t)NRT * NSLAB * 16 * 4);
  float*  per     = (float*) alloc((size_t)BATCH * 4);
  int*    cqpos   = (int*)   alloc((size_t)BATCH * 4);

  kBadMask<<<(NCOL + 7) / 8, 256, 0, stream>>>(lut, cq, badLut, badCq);
  kStats<<<120, 256, 0, stream>>>(lut, badLut, part);
  kStatsReduce<<<1, 256, 0, stream>>>(part, badLut, bw, bb, stats);
  kQn<<<(BATCH + 7) / 8, 256, 0, stream>>>(inp, bw, bb, Qn);
  kW<<<(NCOL + 7) / 8, 256, 0, stream>>>(lut, cq, badLut, badCq, stats, Wm, colBad);
  kRowAux<<<(BATCH + 255) / 256, 256, 0, stream>>>(label, badLut, bpc);
  kGemm<<<NRB * NSLAB, 256, 0, stream>>>(Qn, Wm, colBad, bpc, sPart);
  kTargetRow<<<(BATCH + 7) / 8, 256, 0, stream>>>(Qn, Wm, label, bpc, sPart, per);
  kLossReduce<<<1, 256, 0, stream>>>(per, out);

  hipMemcpyAsync(outLut, lut, (size_t)NPID * FEAT * sizeof(float),
                 hipMemcpyDeviceToDevice, stream);
  hipMemcpyAsync(outCq, cq, (size_t)CQN * FEAT * sizeof(float),
                 hipMemcpyDeviceToDevice, stream);
  kCqScan<<<1, 256, 0, stream>>>(label, cqpos);
  kCqWrite<<<(BATCH + 7) / 8, 256, 0, stream>>>(inp, cqpos, outCq);
  kLutUpdate<<<(BATCH + 7) / 8, 256, 0, stream>>>(inp, label, lut, outLut);
}
